// MelDecoder_55267639165321
// MI455X (gfx1250) — compile-verified, matrix operand order checked by reading, not run on hardware
//
#include <hip/hip_runtime.h>

// Problem constants (match reference)
#define B_    64
#define S_    512
#define D_    512
#define H_    8
#define HD_   64
#define MEL_  80
#define TMAX_ (S_ * 5)   // 2560

typedef __attribute__((ext_vector_type(16))) __bf16        v16bf;
typedef __attribute__((ext_vector_type(8)))  float         v8f;
typedef __attribute__((ext_vector_type(8)))  unsigned int  v8u;

// ---------------- bf16 helpers (stored as ushort) ----------------
__device__ __forceinline__ unsigned short f2bf(float f) {
    unsigned int u = __float_as_uint(f);
    u += 0x7FFFu + ((u >> 16) & 1u);     // round-to-nearest-even
    return (unsigned short)(u >> 16);
}
__device__ __forceinline__ float bf2f(unsigned short h) {
    return __uint_as_float(((unsigned int)h) << 16);
}
__device__ __forceinline__ float sigmoidf_(float x) {
    return 1.0f / (1.0f + __expf(-x));
}

// ---------------- WMMA fragment load ----------------
// ISA 7.12.2, 16-bit A 16x32 (and mirrored B 32x16):
//   lane L: row/col = L&15 ; lanes 16-31 carry K offset +8 (lo half) / +24 (hi half)
//   VGPR 0..3 -> K pairs {0,1}..{6,7}(+hi8), VGPR 4..7 -> K pairs {16,17}..{22,23}(+hi8)
// Source is row-major [rows, stride] bf16 (torch weight layout W[N][K] gives B=W^T for free).
__device__ __forceinline__ v16bf load_frag(const unsigned short* __restrict__ base,
                                           int stride, int row0, int k0, int lane) {
    int r = row0 + (lane & 15);
    const unsigned short* p = base + (long)r * stride + k0 + ((lane >> 4) << 3);
    v8u u;
    u[0] = *(const unsigned int*)(p + 0);
    u[1] = *(const unsigned int*)(p + 2);
    u[2] = *(const unsigned int*)(p + 4);
    u[3] = *(const unsigned int*)(p + 6);
    u[4] = *(const unsigned int*)(p + 16);
    u[5] = *(const unsigned int*)(p + 18);
    u[6] = *(const unsigned int*)(p + 20);
    u[7] = *(const unsigned int*)(p + 22);
    return __builtin_bit_cast(v16bf, u);
}

// ---------------- Generic WMMA GEMM ----------------
// C[M,N] = A[M,K] * W[N,K]^T + bias.  One 16x16 tile per wave, K-loop of bf16 WMMAs.
// Used for: K/V precompute (M=32768), attention out-proj, mel projection.
__global__ void gemm_bf16_wmma(const unsigned short* __restrict__ A,
                               const unsigned short* __restrict__ W,
                               const float* __restrict__ bias,
                               float* __restrict__ outF,
                               unsigned short* __restrict__ outB,
                               int M, int N, int K, long ldc, long ooff) {
    int wave = (int)((blockIdx.x * blockDim.x + threadIdx.x) >> 5);
    int lane = (int)(threadIdx.x & 31);
    int tn = N >> 4;
    int ntile = (M >> 4) * tn;
    if (wave >= ntile) return;                      // wave-uniform: EXEC stays all-1s
    int m0 = (wave / tn) << 4;
    int n0 = (wave % tn) << 4;

    int nc = n0 + (lane & 15);                      // C/D: lane -> column
    float binit = bias ? bias[nc] : 0.0f;
    v8f c;
#pragma unroll
    for (int v = 0; v < 8; ++v) c[v] = binit;

    for (int k0 = 0; k0 < K; k0 += 32) {
        v16bf a = load_frag(A, K, m0, k0, lane);
        v16bf b = load_frag(W, K, n0, k0, lane);
        c = __builtin_amdgcn_wmma_f32_16x16x32_bf16(false, a, false, b,
                                                    (short)0, c, false, false);
    }

    int mhi = (lane >> 4) << 3;                     // C/D: vgpr v -> row v (+8 for lanes 16-31)
    if (outF) {
#pragma unroll
        for (int v = 0; v < 8; ++v)
            outF[ooff + (long)(m0 + mhi + v) * ldc + nc] = c[v];
    } else {
#pragma unroll
        for (int v = 0; v < 8; ++v)
            outB[(long)(m0 + mhi + v) * ldc + nc] = f2bf(c[v]);
    }
}

// ---------------- Fully fused LSTM cell (WMMA gates + in-register nonlinearity) ------
// gates = X*Wih^T + Hin*Whh^T + b_ih + b_hh  (torch gate order i,f,g,o along N=2048)
// Each wave owns one 16-wide j-slice of ALL FOUR gate blocks (1x4 register blocking:
// 1 A fragment + 4 B fragments -> 4 WMMAs per K-chunk), so i,f,g,o for its 16x16
// (b,j) patch live in registers and the cell update is applied in the epilogue.
// Hout must NOT alias X or Hin (ping-pong buffers handle this).  C is owned
// exclusively per-element by one wave -> safe read-modify-write.
__global__ void lstm_cell_wmma(const unsigned short* __restrict__ X,    // [64,512] bf16
                               const unsigned short* __restrict__ Hin,  // [64,512] bf16
                               const unsigned short* __restrict__ Wih,  // [2048,512] bf16
                               const unsigned short* __restrict__ Whh,  // [2048,512] bf16
                               const float* __restrict__ b_ih,          // [2048]
                               const float* __restrict__ b_hh,          // [2048]
                               float* __restrict__ Cst,                 // [64,512] f32 (persistent)
                               unsigned short* __restrict__ Hout) {     // [64,512] bf16
    int wave = (int)((blockIdx.x * blockDim.x + threadIdx.x) >> 5);     // 0..127
    int lane = (int)(threadIdx.x & 31);
    if (wave >= 128) return;                        // wave-uniform
    int j0 = (wave & 31) << 4;                      // j-tile within D
    int m0 = (wave >> 5) << 4;                      // batch tile (0..3)
    int jc = j0 + (lane & 15);

    v8f acc[4];
#pragma unroll
    for (int g = 0; g < 4; ++g) {
        float b = b_ih[g * D_ + jc] + b_hh[g * D_ + jc];
#pragma unroll
        for (int v = 0; v < 8; ++v) acc[g][v] = b;
    }

    // x-path
    for (int k0 = 0; k0 < D_; k0 += 32) {
        v16bf a = load_frag(X, D_, m0, k0, lane);
#pragma unroll
        for (int g = 0; g < 4; ++g) {
            v16bf b = load_frag(Wih, D_, g * D_ + j0, k0, lane);
            acc[g] = __builtin_amdgcn_wmma_f32_16x16x32_bf16(false, a, false, b,
                                                             (short)0, acc[g], false, false);
        }
    }
    // h-path
    for (int k0 = 0; k0 < D_; k0 += 32) {
        v16bf a = load_frag(Hin, D_, m0, k0, lane);
#pragma unroll
        for (int g = 0; g < 4; ++g) {
            v16bf b = load_frag(Whh, D_, g * D_ + j0, k0, lane);
            acc[g] = __builtin_amdgcn_wmma_f32_16x16x32_bf16(false, a, false, b,
                                                             (short)0, acc[g], false, false);
        }
    }

    int mhi = (lane >> 4) << 3;
#pragma unroll
    for (int v = 0; v < 8; ++v) {
        long idx = (long)(m0 + mhi + v) * D_ + jc;
        float si = sigmoidf_(acc[0][v]);
        float sf = sigmoidf_(acc[1][v]);
        float gg = tanhf(acc[2][v]);
        float so = sigmoidf_(acc[3][v]);
        float cn = sf * Cst[idx] + si * gg;
        Cst[idx] = cn;
        Hout[idx] = f2bf(so * tanhf(cn));
    }
}

// ---------------- Attention: fused q-proj + scores + softmax + context ----------------
// One workgroup (256 thr) per (b,h).  h11: bf16 [64,512]; wq: bf16 [512,512] (rows of
// attn_in_w); K/V: bf16 [B*S, 512] with feature index h*64+d.  ctx out: bf16 [64,512].
__global__ void attention_kernel(const unsigned short* __restrict__ h11,
                                 const unsigned short* __restrict__ wq,
                                 const float* __restrict__ bq,
                                 const unsigned short* __restrict__ Kb,
                                 const unsigned short* __restrict__ Vb,
                                 unsigned short* __restrict__ ctx) {
    __shared__ float hs[D_];
    __shared__ float qs[HD_];
    __shared__ float sc[S_];
    __shared__ float red[256];
    int b = (int)(blockIdx.x >> 3);
    int h = (int)(blockIdx.x & 7);
    int t = (int)threadIdx.x;

    // stage h11 row for this batch
    hs[t]       = bf2f(h11[b * D_ + t]);
    hs[t + 256] = bf2f(h11[b * D_ + t + 256]);
    __syncthreads();

    // fused q-projection: q[d] = h11[b,:] . wq[h*64+d,:] + bq ; 4 partials per output
    {
        int d = t >> 2, p = t & 3;
        const unsigned short* wr = wq + (long)(h * HD_ + d) * D_ + p * 128;
        float acc = 0.0f;
#pragma unroll 8
        for (int k = 0; k < 128; ++k) acc += hs[p * 128 + k] * bf2f(wr[k]);
        red[t] = acc;
    }
    __syncthreads();
    if (t < HD_) {
        float s = red[4 * t] + red[4 * t + 1] + red[4 * t + 2] + red[4 * t + 3];
        qs[t] = (s + bq[h * HD_ + t]) * 0.125f;     // 1/sqrt(64)
    }
    __syncthreads();

    // scores
    for (int s = t; s < S_; s += 256) {
        const unsigned short* kp = Kb + ((long)b * S_ + s) * D_ + h * HD_;
        float acc = 0.0f;
#pragma unroll 8
        for (int d = 0; d < HD_; ++d) acc += qs[d] * bf2f(kp[d]);
        sc[s] = acc;
    }
    __syncthreads();

    // max-reduce
    red[t] = fmaxf(sc[t], sc[t + 256]);
    __syncthreads();
    for (int off = 128; off > 0; off >>= 1) {
        if (t < off) red[t] = fmaxf(red[t], red[t + off]);
        __syncthreads();
    }
    float mx = red[0];
    __syncthreads();

    // exp + sum-reduce
    float e0 = __expf(sc[t] - mx);
    float e1 = __expf(sc[t + 256] - mx);
    sc[t] = e0; sc[t + 256] = e1;
    red[t] = e0 + e1;
    __syncthreads();
    for (int off = 128; off > 0; off >>= 1) {
        if (t < off) red[t] += red[t + off];
        __syncthreads();
    }
    float inv = 1.0f / red[0];
    __syncthreads();

    // context: 4 partial accumulators per d (part=t>>6, d=t&63 -> coalesced V reads)
    int part = t >> 6;
    int d    = t & 63;
    const unsigned short* vp = Vb + ((long)b * S_ + part * 128) * D_ + h * HD_ + d;
    float acc = 0.0f;
    for (int s = 0; s < 128; ++s) acc += sc[part * 128 + s] * bf2f(vp[(long)s * D_]);
    red[t] = acc;
    __syncthreads();
    if (t < 64) {
        float r = (red[t] + red[t + 64] + red[t + 128] + red[t + 192]) * inv;
        ctx[b * D_ + h * HD_ + t] = f2bf(r);
    }
}

// ---------------- small utility kernels ----------------
__global__ void f32_to_bf16(const float* __restrict__ in, unsigned short* __restrict__ out, int n) {
    int i = (int)(blockIdx.x * blockDim.x + threadIdx.x);
    if (i < n) out[i] = f2bf(in[i]);
}

__global__ void init_state(float* c0, float* c1, float* c2,
                           unsigned short* h0, unsigned short* h1, unsigned short* h2) {
    int i = (int)(blockIdx.x * blockDim.x + threadIdx.x);   // 64*512 threads exactly
    c0[i] = 0.0f; c1[i] = 0.0f; c2[i] = 0.0f;
    h0[i] = 0;    h1[i] = 0;    h2[i] = 0;
}

// ---------------- host-side launch helpers ----------------
static inline void launch_gemm(hipStream_t st,
                               const unsigned short* A, const unsigned short* W,
                               const float* b1, float* oF, unsigned short* oB,
                               int M, int N, int K, long ldc, long ooff) {
    int tiles  = (M >> 4) * (N >> 4);
    int blocks = (tiles + 7) / 8;                 // 8 waves / 256-thread block (wave32)
    gemm_bf16_wmma<<<blocks, 256, 0, st>>>(A, W, b1, oF, oB, M, N, K, ldc, ooff);
}

static inline void launch_conv(hipStream_t st, const float* in, unsigned short* out, int n) {
    f32_to_bf16<<<(n + 255) / 256, 256, 0, st>>>(in, out, n);
}

extern "C" void kernel_launch(void* const* d_in, const int* in_sizes, int n_in,
                              void* d_out, int out_size, void* d_ws, size_t ws_size,
                              hipStream_t stream) {
    (void)in_sizes; (void)n_in; (void)out_size; (void)ws_size;

    // ---- inputs (setup_inputs order) ----
    const float* enc        = (const float*)d_in[0];   // [64,512,512]
    const float* attn_in_w  = (const float*)d_in[1];   // [1536,512]
    const float* attn_in_b  = (const float*)d_in[2];   // [1536]
    const float* attn_out_w = (const float*)d_in[3];   // [512,512]
    const float* attn_out_b = (const float*)d_in[4];   // [512]
    const float* w_ih0 = (const float*)d_in[5],  *w_hh0 = (const float*)d_in[6];
    const float* b_ih0 = (const float*)d_in[7],  *b_hh0 = (const float*)d_in[8];
    const float* w_ih1 = (const float*)d_in[9],  *w_hh1 = (const float*)d_in[10];
    const float* b_ih1 = (const float*)d_in[11], *b_hh1 = (const float*)d_in[12];
    const float* w_ih2 = (const float*)d_in[13], *w_hh2 = (const float*)d_in[14];
    const float* b_ih2 = (const float*)d_in[15], *b_hh2 = (const float*)d_in[16];
    const float* proj_w = (const float*)d_in[17];      // [80,512]
    const float* proj_b = (const float*)d_in[18];      // [80]
    float* out = (float*)d_out;                        // [64, 2560, 80]

    // ---- workspace layout (256B aligned slices) ----
    size_t off = 0;
    auto ws = [&](size_t bytes) -> void* {
        void* p = (char*)d_ws + off;
        off += (bytes + 255) & ~(size_t)255;
        return p;
    };
    const long BS = (long)B_ * S_;                                   // 32768
    unsigned short* enc_bf = (unsigned short*)ws(BS * D_ * 2);       // 32 MB
    unsigned short* Kbf    = (unsigned short*)ws(BS * D_ * 2);       // 32 MB (L2-resident)
    unsigned short* Vbf    = (unsigned short*)ws(BS * D_ * 2);       // 32 MB (L2-resident)
    unsigned short* win_bf = (unsigned short*)ws(1536L * 512 * 2);   // wq|wk|wv
    unsigned short* wout_bf= (unsigned short*)ws(512L * 512 * 2);
    unsigned short* wih0_bf= (unsigned short*)ws(2048L * 512 * 2);
    unsigned short* whh0_bf= (unsigned short*)ws(2048L * 512 * 2);
    unsigned short* wih1_bf= (unsigned short*)ws(2048L * 512 * 2);
    unsigned short* whh1_bf= (unsigned short*)ws(2048L * 512 * 2);
    unsigned short* wih2_bf= (unsigned short*)ws(2048L * 512 * 2);
    unsigned short* whh2_bf= (unsigned short*)ws(2048L * 512 * 2);
    unsigned short* wprj_bf= (unsigned short*)ws(80L * 512 * 2);
    float*          c10    = (float*)ws(64L * 512 * 4);
    float*          c11    = (float*)ws(64L * 512 * 4);
    float*          c2     = (float*)ws(64L * 512 * 4);
    // ping-pong hidden states (fused cell reads full H while writing -> double buffer)
    unsigned short* h10p[2] = { (unsigned short*)ws(64L * 512 * 2),
                                (unsigned short*)ws(64L * 512 * 2) };
    unsigned short* h11p[2] = { (unsigned short*)ws(64L * 512 * 2),
                                (unsigned short*)ws(64L * 512 * 2) };
    unsigned short* h2p[2]  = { (unsigned short*)ws(64L * 512 * 2),
                                (unsigned short*)ws(64L * 512 * 2) };
    unsigned short* ctxb  = (unsigned short*)ws(64L * 512 * 2);
    unsigned short* attnb = (unsigned short*)ws(64L * 512 * 2);

    // ---- one-time per call: convert weights/activations to bf16 ----
    launch_conv(stream, enc,        enc_bf,  (int)(BS * D_));
    launch_conv(stream, attn_in_w,  win_bf,  1536 * 512);
    launch_conv(stream, attn_out_w, wout_bf, 512 * 512);
    launch_conv(stream, w_ih0, wih0_bf, 2048 * 512);
    launch_conv(stream, w_hh0, whh0_bf, 2048 * 512);
    launch_conv(stream, w_ih1, wih1_bf, 2048 * 512);
    launch_conv(stream, w_hh1, whh1_bf, 2048 * 512);
    launch_conv(stream, w_ih2, wih2_bf, 2048 * 512);
    launch_conv(stream, w_hh2, whh2_bf, 2048 * 512);
    launch_conv(stream, proj_w, wprj_bf, 80 * 512);

    // ---- K / V precompute over encoder memory (big WMMA GEMMs, bf16 out) ----
    launch_gemm(stream, enc_bf, win_bf + (long)512 * 512,  attn_in_b + 512,
                nullptr, Kbf, (int)BS, 512, 512, 512, 0);
    launch_gemm(stream, enc_bf, win_bf + (long)1024 * 512, attn_in_b + 1024,
                nullptr, Vbf, (int)BS, 512, 512, 512, 0);

    // ---- zero states (parity-0 buffers are the t=0 inputs) ----
    init_state<<<(64 * 512) / 256, 256, 0, stream>>>(c10, c11, c2,
                                                     h10p[0], h11p[0], h2p[0]);

    // ---- sequential decode loop: 6 fused kernels per step ----
    const long out_ldc = (long)TMAX_ * MEL_;
    for (int t = 0; t < TMAX_; ++t) {
        int p = t & 1, q = p ^ 1;

        // LSTM1 layer 0: x = dec_in = h2[p] (prev step's h2; zeros at t=0)
        lstm_cell_wmma<<<16, 256, 0, stream>>>(h2p[p], h10p[p], wih0_bf, whh0_bf,
                                               b_ih0, b_hh0, c10, h10p[q]);
        // LSTM1 layer 1: x = h10[q]
        lstm_cell_wmma<<<16, 256, 0, stream>>>(h10p[q], h11p[p], wih1_bf, whh1_bf,
                                               b_ih1, b_hh1, c11, h11p[q]);
        // attention (q-proj fused) over encoder K/V
        attention_kernel<<<B_ * H_, 256, 0, stream>>>(h11p[q], win_bf, attn_in_b,
                                                      Kbf, Vbf, ctxb);
        // attention output projection (WMMA, bf16 out)
        launch_gemm(stream, ctxb, wout_bf, attn_out_b,
                    nullptr, attnb, 64, 512, 512, 512, 0);
        // LSTM2: x = attn_out
        lstm_cell_wmma<<<16, 256, 0, stream>>>(attnb, h2p[p], wih2_bf, whh2_bf,
                                               b_ih2, b_hh2, c2, h2p[q]);
        // mel projection -> out[:, t, :]
        launch_gemm(stream, h2p[q], wprj_bf, proj_b,
                    out, nullptr, 64, MEL_, 512, out_ldc, (long)t * MEL_);
    }
}
